// Detail_Recovery_13477607375314
// MI455X (gfx1250) — matrix operand order repaired and checked by reading, not compiled, not run on hardware
//
#include <hip/hip_runtime.h>

// ---------------------------------------------------------------------------
// CDNA5 (gfx1250) implementation.  All GEMM-shaped work goes through
// v_wmma_f32_16x16x32_bf16 (wave32, 16x16 tiles, K=32 per issue).
// Intermediates are pixel-major bf16 ([hw][32]) so channel is contiguous:
// A-frags are two aligned 16B chunks, LDS staging is a vector copy, and
// stores are contiguous runs.  Weights are pre-swizzled to fragment layout
// (one aligned 32B load per B-frag) and reused across 4 M-tiles per wave.
// ---------------------------------------------------------------------------

typedef __attribute__((ext_vector_type(16))) __bf16 v16bf;
typedef __attribute__((ext_vector_type(8)))  __bf16 v8bf;
typedef __attribute__((ext_vector_type(8)))  float  v8f;

__device__ __forceinline__ v8f wmma_bf16(v16bf a, v16bf b, v8f c) {
  // D = A(16x32) * B(32x16) + C, f32 accumulate
  return __builtin_amdgcn_wmma_f32_16x16x32_bf16(
      /*neg_a=*/false, a, /*neg_b=*/false, b,
      /*c_mod=*/(short)0, c, /*reuse_a=*/false, /*reuse_b=*/false);
}

__device__ __forceinline__ v16bf mk16(v8bf lo, v8bf hi) {
  return __builtin_shufflevector(lo, hi, 0, 1, 2, 3, 4, 5, 6, 7,
                                 8, 9, 10, 11, 12, 13, 14, 15);
}

// A-matrix 16x32 bf16: lane = m(0..15) + 16*kh.
//   elems 0..7  <-> k = kh*16 + 0..7       (16B chunk)
//   elems 8..15 <-> k = kh*16 + 8..15  (16B chunk)
// B-matrix 32x16 bf16: lane = n(0..15)+16*kh, elem e <-> k = kh*16+e (32B chunk)

#define HW 16384          // 128*128
#define NPIX 65536        // 4*128*128

// ---------------------------------------------------------------------------
// Prep kernels: weight conversion / swizzle (run every launch; tiny).
// ---------------------------------------------------------------------------
__global__ __launch_bounds__(256) void k_cvt_bf16(const float* __restrict__ src,
                                                  __bf16* __restrict__ dst, int n) {
  int i = blockIdx.x * 256 + threadIdx.x;
  if (i < n) dst[i] = (__bf16)src[i];
}

// w3 (NH,6,O=32,I=32,3,3) fp32 -> wf3 [hc(48)][off(9)][nt(2)][lane(32)][e(16)] bf16
__global__ __launch_bounds__(256) void k_swz_w3(const float* __restrict__ w3,
                                                __bf16* __restrict__ wf3) {
  int i = blockIdx.x * 256 + threadIdx.x;
  if (i >= 48 * 9 * 2 * 32 * 16) return;
  int e    = i & 15;
  int lane = (i >> 4) & 31;
  int nt   = (i >> 9) & 1;
  int off  = (i >> 10) % 9;
  int hc   = (i >> 10) / 9;
  int kh   = lane >> 4;
  int n    = (lane & 15) + nt * 16;
  int k    = kh * 16 + e;
  wf3[i] = (__bf16)w3[(((size_t)hc * 32 + n) * 32 + k) * 9 + off];
}

// ---------------------------------------------------------------------------
// Kernel 1: qp = relu(affine(conv1x1(res)))   GEMM M=NPIX, K=256, N=32
// Output PIXEL-MAJOR: qp_t[b][hw][32].
// ---------------------------------------------------------------------------
__global__ __launch_bounds__(128) void k_qp(const float* __restrict__ res,
                                            const __bf16* __restrict__ wv_bf,
                                            const float* __restrict__ wv_s,
                                            const float* __restrict__ wv_b,
                                            float* __restrict__ qp_t) {
  __shared__ __bf16 sA[64][264];   // 64 px x 256 ch (+8 pad)
  __shared__ float  sO[32][68];    // 32 ch x 64 px (+4 pad)

  const int p0  = blockIdx.x * 64;
  const int bat = p0 >> 14;
  const int hw0 = p0 & (HW - 1);
  const float* rb = res + (size_t)bat * 256 * HW + hw0;
  const int t = threadIdx.x;

  for (int i = t; i < 64 * 256; i += 128) {
    int pix = i & 63, ch = i >> 6;
    sA[pix][ch] = (__bf16)rb[(size_t)ch * HW + pix];
  }
  __syncthreads();

  const int lane = t & 31;
  const int m16 = lane & 15, kh = lane >> 4;
  const int pixBase = (t >> 5) * 16;

  v8f acc0 = {}, acc1 = {};
#pragma unroll
  for (int ks = 0; ks < 8; ++ks) {
    v8bf lo = *(const v8bf*)&sA[pixBase + m16][ks * 32 + kh * 16];
    v8bf hi = *(const v8bf*)&sA[pixBase + m16][ks * 32 + kh * 16 + 8];
    v16bf afr = mk16(lo, hi);
    v16bf b0 = *(const v16bf*)&wv_bf[(size_t)m16 * 256 + ks * 32 + kh * 16];
    v16bf b1 = *(const v16bf*)&wv_bf[(size_t)(m16 + 16) * 256 + ks * 32 + kh * 16];
    acc0 = wmma_bf16(afr, b0, acc0);
    acc1 = wmma_bf16(afr, b1, acc1);
  }
#pragma unroll
  for (int r = 0; r < 8; ++r) {
    int m = r + 8 * kh;
    sO[m16][pixBase + m]      = fmaxf(fmaf(acc0[r], wv_s[m16],      wv_b[m16]),      0.f);
    sO[m16 + 16][pixBase + m] = fmaxf(fmaf(acc1[r], wv_s[m16 + 16], wv_b[m16 + 16]), 0.f);
  }
  __syncthreads();
  float* ob = qp_t + (size_t)p0 * 32;
  for (int i = t; i < 64 * 32; i += 128) {
    int pix = i >> 5, ch = i & 31;
    ob[i] = sO[ch][pix];
  }
}

// ---------------------------------------------------------------------------
// Kernel 2: point-aware local attention (dilated 5x5, stride-2 offsets).
// ---------------------------------------------------------------------------
__global__ __launch_bounds__(256) void k_pattn(const float* __restrict__ qp_t,
                                               float* __restrict__ pc_t) {
  const int p = blockIdx.x * 256 + threadIdx.x;
  const int bat = p >> 14, hw = p & (HW - 1);
  const int h = hw >> 7, w = hw & 127;
  const float* qt = qp_t + (size_t)bat * HW * 32;

  float q[32];
#pragma unroll
  for (int c = 0; c < 32; ++c) q[c] = qt[(size_t)hw * 32 + c];
  __builtin_prefetch(&qt[(size_t)(hw + 128) * 32], 0, 1);

  float sc[25];
  float mx = 0.f;
#pragma unroll
  for (int k = 0; k < 25; ++k) {
    int di = (k / 5) * 2 - 4, dj = (k % 5) * 2 - 4;
    int hh = h + di, ww = w + dj;
    float s = 0.f;
    if (k != 12 && hh >= 0 && hh < 128 && ww >= 0 && ww < 128) {
      const float* nb = qt + (size_t)(hh * 128 + ww) * 32;
#pragma unroll
      for (int c = 0; c < 32; ++c) s = fmaf(q[c], nb[c], s);
    }
    sc[k] = s;
    mx = fmaxf(mx, s);
  }
  float sum = 0.f;
#pragma unroll
  for (int k = 0; k < 25; ++k) { sc[k] = __expf(sc[k] - mx); sum += sc[k]; }
  const float inv = 1.f / sum;

  float acc[32];
#pragma unroll
  for (int c = 0; c < 32; ++c) acc[c] = q[c];
#pragma unroll
  for (int k = 0; k < 25; ++k) {
    int di = (k / 5) * 2 - 4, dj = (k % 5) * 2 - 4;
    int hh = h + di, ww = w + dj;
    if (hh < 0 || hh >= 128 || ww < 0 || ww >= 128) continue;  // padded q25 == 0
    float wk = sc[k] * inv;
    const float* nb = qt + (size_t)(hh * 128 + ww) * 32;
#pragma unroll
    for (int c = 0; c < 32; ++c) acc[c] = fmaf(wk, nb[c], acc[c]);
  }
  float* ot = pc_t + (size_t)p * 32;
#pragma unroll
  for (int c = 0; c < 32; ++c) ot[c] = acc[c];
}

// ---------------------------------------------------------------------------
// Kernel 3: class attention + per-head norm/residual -> xi bf16 PIXEL-MAJOR
// xi layout: [head][b][hw][32]
// ---------------------------------------------------------------------------
__global__ __launch_bounds__(256) void k_cattn(const float* __restrict__ pc_t,
                                               const float* __restrict__ attn,
                                               const float* __restrict__ cn_s,
                                               const float* __restrict__ cn_b,
                                               __bf16* __restrict__ xi) {
  const int p = blockIdx.x * 256 + threadIdx.x;
  const int bat = p >> 14, hw = p & (HW - 1);
  const float* pcb = pc_t + (size_t)p * 32;
  const float* ab  = attn + (size_t)bat * 256 * HW + hw;

  float pv[32];
#pragma unroll
  for (int c = 0; c < 32; ++c) pv[c] = pcb[c];

  float s[8];
  float mx = -1e30f;
#pragma unroll
  for (int n = 0; n < 8; ++n) {
    float d = 0.f;
#pragma unroll
    for (int c = 0; c < 32; ++c) d = fmaf(pv[c], ab[(size_t)(n * 32 + c) * HW], d);
    s[n] = d;
    mx = fmaxf(mx, d);
  }
  float sum = 0.f;
#pragma unroll
  for (int n = 0; n < 8; ++n) { s[n] = __expf(s[n] - mx); sum += s[n]; }
  const float inv = 1.f / sum;

#pragma unroll
  for (int n = 0; n < 8; ++n) {
    const float cs = s[n] * inv;
    __bf16* xo = xi + ((size_t)(n * 4 + bat) * HW + hw) * 32;
#pragma unroll
    for (int c = 0; c < 32; ++c) {
      float normed = fmaf(cs * pv[c], cn_s[n * 32 + c], cn_b[n * 32 + c]);
      xo[c] = (__bf16)(ab[(size_t)(n * 32 + c) * HW] + normed);
    }
  }
}

// ---------------------------------------------------------------------------
// Kernel 4 (templated on branch count): multi-branch 3x3 conv stage
//   out = sum_{br} relu6(conv3x3(in, w3[head][convBase+br]) * s3 + b3)
// Block = 8 waves covering 4 rows x 128 cols; each wave: 64 px (4 M-tiles),
// so every B-frag load feeds 4 WMMAs.  In/out pixel-major bf16.
// ---------------------------------------------------------------------------
template <int NB>
__global__ __launch_bounds__(256) void k_msconv3(const __bf16* __restrict__ in,
                                                 const __bf16* __restrict__ wf3,
                                                 const float* __restrict__ s3,
                                                 const float* __restrict__ b3,
                                                 __bf16* __restrict__ out,
                                                 int convBase) {
  __shared__ __bf16 sIn[6][130][40];   // rows h0-1..h0+4, col 0..129, ch (+8 pad)

  const int head = blockIdx.y;
  const int bat = blockIdx.x >> 5;
  const int h0 = (blockIdx.x & 31) * 4;
  const int t = threadIdx.x;
  const __bf16* ib = in + ((size_t)(head * 4 + bat) * HW) * 32;

  // stage 6 rows x 130 cols x 32 ch, 8 bf16 per chunk (vectorized copy)
  for (int j = t; j < 6 * 130 * 4; j += 256) {
    int ch8 = j & 3;
    int col = (j >> 2) % 130;
    int dyb = (j >> 2) / 130;
    int hh = h0 - 1 + dyb, ww = col - 1;
    v8bf v = {};
    if (hh >= 0 && hh < 128 && ww >= 0 && ww < 128)
      v = *(const v8bf*)&ib[(size_t)(hh * 128 + ww) * 32 + ch8 * 8];
    *(v8bf*)&sIn[dyb][col][ch8 * 8] = v;
  }
  __syncthreads();

  const int wave = t >> 5, lane = t & 31;
  const int m16 = lane & 15, kh = lane >> 4;
  const int rowIn = wave >> 1;          // 0..3 (row within block)
  const int w0 = (wave & 1) * 64;       // column half
  const size_t hcBase = (size_t)(head * 6 + convBase);

  v8f sum[4][2] = {};
#pragma unroll
  for (int br = 0; br < NB; ++br) {
    v8f acc[4][2] = {};
#pragma unroll
    for (int dy = 0; dy < 3; ++dy)
#pragma unroll
      for (int dx = 0; dx < 3; ++dx) {
        const int off = dy * 3 + dx;
        const __bf16* wp =
            wf3 + ((((hcBase + br) * 9 + off) * 2) * 32 + lane) * 16;
        v16bf b0 = *(const v16bf*)wp;
        v16bf b1 = *(const v16bf*)(wp + 512);   // nt=1 plane (32 lanes * 16)
#pragma unroll
        for (int mt = 0; mt < 4; ++mt) {
          const int col = w0 + mt * 16 + m16 + dx;
          v8bf lo = *(const v8bf*)&sIn[rowIn + dy][col][kh * 16];
          v8bf hi = *(const v8bf*)&sIn[rowIn + dy][col][kh * 16 + 8];
          v16bf afr = mk16(lo, hi);
          acc[mt][0] = wmma_bf16(afr, b0, acc[mt][0]);
          acc[mt][1] = wmma_bf16(afr, b1, acc[mt][1]);
        }
      }
    // per-branch affine + relu6, accumulate
    const float* sB = s3 + ((size_t)head * 6 + convBase + br) * 32;
    const float* bB = b3 + ((size_t)head * 6 + convBase + br) * 32;
#pragma unroll
    for (int nt = 0; nt < 2; ++nt) {
      const float sc = sB[m16 + nt * 16], bc = bB[m16 + nt * 16];
#pragma unroll
      for (int mt = 0; mt < 4; ++mt)
#pragma unroll
        for (int r = 0; r < 8; ++r) {
          float v = fmaf(acc[mt][nt][r], sc, bc);
          sum[mt][nt][r] += fminf(fmaxf(v, 0.f), 6.f);
        }
    }
  }

  // store pixel-major: contiguous 16 channels per half-wave
  __bf16* ob = out + ((size_t)(head * 4 + bat) * HW + (h0 + rowIn) * 128) * 32;
#pragma unroll
  for (int mt = 0; mt < 4; ++mt)
#pragma unroll
    for (int nt = 0; nt < 2; ++nt)
#pragma unroll
      for (int r = 0; r < 8; ++r) {
        int pix = w0 + mt * 16 + r + 8 * kh;
        int n = m16 + nt * 16;
        ob[(size_t)pix * 32 + n] = (__bf16)sum[mt][nt][r];
      }
}

// ---------------------------------------------------------------------------
// Kernel 5: cat-conv 1x1 (K=96 -> N=32) + residual + relu -> NCHW fp32 out.
// A-frags straight from global (pixel-major sources are channel-contiguous).
// ---------------------------------------------------------------------------
__global__ __launch_bounds__(256) void k_cat(const __bf16* __restrict__ x112,
                                             const __bf16* __restrict__ x223,
                                             const __bf16* __restrict__ x33,
                                             const __bf16* __restrict__ xi,
                                             const __bf16* __restrict__ wcat_bf,
                                             const float* __restrict__ scat,
                                             const float* __restrict__ bcat,
                                             float* __restrict__ out) {
  __shared__ float sO[32][132];

  const int head = blockIdx.y;
  const int bh = blockIdx.x;
  const int bat = bh >> 7, h = bh & 127;
  const int t = threadIdx.x;
  const size_t rowPix = (size_t)(head * 4 + bat) * HW + (size_t)h * 128;
  const __bf16* srcs[3] = {x112 + rowPix * 32, x223 + rowPix * 32, x33 + rowPix * 32};

  const int wave = t >> 5, lane = t & 31;
  const int m16 = lane & 15, kh = lane >> 4;
  const int w0 = wave * 16;

  v8f acc0 = {}, acc1 = {};
#pragma unroll
  for (int src = 0; src < 3; ++src) {
    const __bf16* sp = srcs[src] + (size_t)(w0 + m16) * 32;
    v8bf lo = *(const v8bf*)&sp[kh * 16];
    v8bf hi = *(const v8bf*)&sp[kh * 16 + 8];
    v16bf afr = mk16(lo, hi);
    v16bf b0 = *(const v16bf*)&wcat_bf[((size_t)head * 32 + m16) * 96 + src * 32 + kh * 16];
    v16bf b1 = *(const v16bf*)&wcat_bf[((size_t)head * 32 + m16 + 16) * 96 + src * 32 + kh * 16];
    acc0 = wmma_bf16(afr, b0, acc0);
    acc1 = wmma_bf16(afr, b1, acc1);
  }

#pragma unroll
  for (int nt = 0; nt < 2; ++nt) {
    const int n = m16 + nt * 16;
    const float sc = scat[head * 32 + n], bc = bcat[head * 32 + n];
#pragma unroll
    for (int r = 0; r < 8; ++r) {
      int m = r + 8 * kh;
      float xv = (float)xi[(rowPix + w0 + m) * 32 + n];
      float v = fmaf((nt == 0 ? acc0[r] : acc1[r]), sc, bc) + xv;
      sO[n][w0 + m] = fmaxf(v, 0.f);
    }
  }
  __syncthreads();

  float* ob = out + ((size_t)bat * 256 + head * 32) * HW + (size_t)h * 128;
  for (int i = t; i < 32 * 128; i += 256) {
    int ch = i >> 7, ww = i & 127;
    ob[(size_t)ch * HW + ww] = sO[ch][ww];
  }
}

// ---------------------------------------------------------------------------
// Host launcher
// ---------------------------------------------------------------------------
extern "C" void kernel_launch(void* const* d_in, const int* in_sizes, int n_in,
                              void* d_out, int out_size, void* d_ws, size_t ws_size,
                              hipStream_t stream) {
  const float* res   = (const float*)d_in[0];
  const float* attn  = (const float*)d_in[1];
  const float* wv_w  = (const float*)d_in[2];
  const float* wv_s  = (const float*)d_in[3];
  const float* wv_b  = (const float*)d_in[4];
  const float* cn_s  = (const float*)d_in[5];
  const float* cn_b  = (const float*)d_in[6];
  const float* w3    = (const float*)d_in[7];
  const float* s3    = (const float*)d_in[8];
  const float* b3    = (const float*)d_in[9];
  const float* wcat  = (const float*)d_in[10];
  const float* scat  = (const float*)d_in[11];
  const float* bcat  = (const float*)d_in[12];
  float* out = (float*)d_out;

  char* ws = (char*)d_ws;
  const size_t SZ_F32 = (size_t)NPIX * 32 * sizeof(float);      // 8 MiB
  const size_t SZ_BF  = (size_t)8 * NPIX * 32 * sizeof(__bf16); // 32 MiB
  float*  qp_t = (float*)(ws);
  float*  pc_t = (float*)(ws + SZ_F32);
  __bf16* xi   = (__bf16*)(ws + 2 * SZ_F32);
  __bf16* x112 = (__bf16*)(ws + 2 * SZ_F32 + SZ_BF);
  __bf16* x223 = (__bf16*)(ws + 2 * SZ_F32 + 2 * SZ_BF);
  __bf16* x33  = (__bf16*)(ws + 2 * SZ_F32 + 3 * SZ_BF);
  char*   wtop = ws + 2 * SZ_F32 + 4 * SZ_BF;
  __bf16* wv_bf   = (__bf16*)(wtop);                 // 32*256       = 8192
  __bf16* wcat_bf = (__bf16*)(wtop + 16384);         // 8*32*96      = 24576
  __bf16* wf3     = (__bf16*)(wtop + 65536);         // 48*9*2*32*16 = 442368

  // 0) weight prep: bf16 convert + fragment swizzle
  k_cvt_bf16<<<(8192 + 255) / 256, 256, 0, stream>>>(wv_w, wv_bf, 8192);
  k_cvt_bf16<<<(24576 + 255) / 256, 256, 0, stream>>>(wcat, wcat_bf, 24576);
  k_swz_w3<<<(442368 + 255) / 256, 256, 0, stream>>>(w3, wf3);

  // 1) qp GEMM (WMMA bf16) -> pixel-major qp_t
  k_qp<<<NPIX / 64, 128, 0, stream>>>(res, wv_bf, wv_s, wv_b, qp_t);
  // 2) point-aware local attention (VALU, contiguous neighbor reads)
  k_pattn<<<NPIX / 256, 256, 0, stream>>>(qp_t, pc_t);
  // 3) class attention + per-head residual -> xi (bf16 pixel-major)
  k_cattn<<<NPIX / 256, 256, 0, stream>>>(pc_t, attn, cn_s, cn_b, xi);
  // 4) MSConv stages (WMMA implicit GEMM, 4 rows per block, 64 px per wave)
  dim3 cgrid(128, 8);  // (b*rowgroups, heads)
  k_msconv3<3><<<cgrid, 256, 0, stream>>>(xi,   wf3, s3, b3, x112, 0);
  k_msconv3<2><<<cgrid, 256, 0, stream>>>(x112, wf3, s3, b3, x223, 3);
  k_msconv3<1><<<cgrid, 256, 0, stream>>>(x223, wf3, s3, b3, x33,  5);
  // 5) cat conv + residual + relu -> output (WMMA)
  dim3 ggrid(512, 8);  // (b*h rows, heads)
  k_cat<<<ggrid, 256, 0, stream>>>(x112, x223, x33, xi, wcat_bf, scat, bcat, out);
}
